// Multiheadattention_11854109737408
// MI455X (gfx1250) — matrix operand order repaired and checked by reading, not compiled, not run on hardware
//
#include <hip/hip_runtime.h>
#include <hip/hip_bf16.h>

// MHA forward for B=4, S=2048, D=1024, H=16, hd=64 on gfx1250 (MI455X).
// bf16 WMMA everywhere, f32 accumulate. Global->LDS async double-buffering.

typedef __bf16 bf16;
typedef __attribute__((ext_vector_type(16))) __bf16 v16bf;
typedef __attribute__((ext_vector_type(8)))  __bf16 v8bf;
typedef __attribute__((ext_vector_type(4)))  __bf16 v4bf;
typedef __attribute__((ext_vector_type(8)))  float  v8f;
typedef __attribute__((ext_vector_type(4)))  int    v4i;

#define DEV static __device__ __forceinline__
#define GLOBAL_AS __attribute__((address_space(1)))
#define LDS_AS    __attribute__((address_space(3)))

#ifndef __has_builtin
#define __has_builtin(x) 0
#endif

// ---- async global -> LDS copy (ASYNCcnt path), with graceful fallback ----
DEV void cp_async_b128(bf16* l, const bf16* g) {
#if __has_builtin(__builtin_amdgcn_global_load_async_to_lds_b128)
  __builtin_amdgcn_global_load_async_to_lds_b128(
      (GLOBAL_AS v4i*)(void*)g, (LDS_AS v4i*)(void*)l, 0, 0);
#else
  *(v8bf*)l = *(const v8bf*)g;   // sync fallback: global_load + ds_store
#endif
}

DEV void wait_async() {
#if __has_builtin(__builtin_amdgcn_s_wait_asynccnt)
  __builtin_amdgcn_s_wait_asynccnt(0);
#else
  asm volatile("s_wait_asynccnt 0" ::: "memory");
#endif
}

DEV void syncpipe() {
  wait_async();
  __syncthreads();
}

// ---------------- WMMA fragment helpers (CDNA5 16-bit layouts) ----------------
// A 16x32 (MxK): lane L (M=L&15), hi=L>>4; elems 0..7 -> K=kb+hi*8+i,
// elems 8..15 -> K=kb+16+hi*8+(i-8).
DEV v16bf load_a16(const bf16* p, int ld, int row, int kb, int lane) {
  const int r  = row + (lane & 15);
  const int hi = lane >> 4;
  const bf16* base = p + (size_t)r * ld + kb + hi * 8;
  union { v16bf v; v8bf h[2]; } u;
  u.h[0] = *(const v8bf*)(base);
  u.h[1] = *(const v8bf*)(base + 16);
  return u.v;
}

// B 32x16 (KxN) from transposed storage Bt[N][K]: lane L (N=L&15),
// elems 0..7 -> K = kb + hi*8 + i, elems 8..15 -> K = kb + 16 + hi*8 + (i-8).
DEV v16bf load_bt16(const bf16* p, int ld, int col, int kb, int lane) {
  const int c  = col + (lane & 15);
  const int hi = lane >> 4;
  const bf16* base = p + (size_t)c * ld + kb + hi * 8;
  union { v16bf v; v8bf h[2]; } u;
  u.h[0] = *(const v8bf*)(base);
  u.h[1] = *(const v8bf*)(base + 16);
  return u.v;
}

DEV v8f wmma_bf16(v16bf a, v16bf b, v8f c) {
  return __builtin_amdgcn_wmma_f32_16x16x32_bf16(
      false, a, false, b, (short)0, c, false, false);
}

// 16-lane half reductions (rows replicated per wave32 half).
DEV float halfmax(float v) {
  v = fmaxf(v, __shfl_xor(v, 1, 32));
  v = fmaxf(v, __shfl_xor(v, 2, 32));
  v = fmaxf(v, __shfl_xor(v, 4, 32));
  v = fmaxf(v, __shfl_xor(v, 8, 32));
  return v;
}
DEV float halfsum(float v) {
  v += __shfl_xor(v, 1, 32);
  v += __shfl_xor(v, 2, 32);
  v += __shfl_xor(v, 4, 32);
  v += __shfl_xor(v, 8, 32);
  return v;
}

// ---------------- precision conversion / weight transpose ----------------
__global__ __launch_bounds__(256) void cvt_f32_bf16(
    const float* __restrict__ in, bf16* __restrict__ out, int n) {
  int i = (blockIdx.x * 256 + threadIdx.x) * 4;
  if (i + 3 < n) {
    float4 f = *(const float4*)(in + i);
    v4bf o;
    o[0] = (bf16)f.x; o[1] = (bf16)f.y; o[2] = (bf16)f.z; o[3] = (bf16)f.w;
    *(v4bf*)(out + i) = o;
  }
}

// wt[n*K + k] = (bf16) w[k*N + n]
__global__ __launch_bounds__(256) void transpose_cvt(
    const float* __restrict__ w, bf16* __restrict__ wt, int K, int N) {
  int t = blockIdx.x * 256 + threadIdx.x;
  if (t < K * N) {
    int n = t / K;
    int k = t - n * K;
    wt[t] = (bf16)w[(size_t)k * N + n];
  }
}

// ---------------- shared GEMM core ----------------
// Block tile 64(M) x 256(N), K-tile 64, double-buffered async LDS staging.
// 8 waves arranged 2x4; each wave -> 32x64 via 2x4 grid of 16x16 WMMAs.
DEV void gemm_core(const bf16* __restrict__ A, const bf16* __restrict__ Bt,
                   int row0g, int n0g, int Kdim,
                   bf16* lA, bf16* lB, v8f (&acc)[2][4]) {
  const int tid  = threadIdx.x;
  const int lane = tid & 31;
  const int wm   = (tid >> 5) >> 2;    // 0..1
  const int wn   = (tid >> 5) & 3;     // 0..3

  auto stage = [&](int buf, int k) {
    bf16* la = lA + buf * (64 * 64);
    bf16* lb = lB + buf * (256 * 64);
#pragma unroll
    for (int i = 0; i < 2; ++i) {      // A tile: 64 rows x 128B
      int idx = i * 256 + tid;
      int r = idx >> 3, c = (idx & 7) * 8;
      cp_async_b128(la + r * 64 + c, A + (size_t)(row0g + r) * 1024 + k + c);
    }
#pragma unroll
    for (int i = 0; i < 8; ++i) {      // Bt tile: 256 rows x 128B
      int idx = i * 256 + tid;
      int r = idx >> 3, c = (idx & 7) * 8;
      cp_async_b128(lb + r * 64 + c, Bt + (size_t)(n0g + r) * 1024 + k + c);
    }
  };

#pragma unroll
  for (int mi = 0; mi < 2; ++mi)
#pragma unroll
    for (int j = 0; j < 4; ++j) acc[mi][j] = {};

  stage(0, 0);
  syncpipe();
  for (int k = 0; k < Kdim; k += 64) {
    const int cur = (k >> 6) & 1;
    if (k + 64 < Kdim) stage(cur ^ 1, k + 64);
    const bf16* la = lA + cur * (64 * 64);
    const bf16* lb = lB + cur * (256 * 64);
#pragma unroll
    for (int ks = 0; ks < 64; ks += 32) {
      v16bf a0 = load_a16(la, 64, wm * 32, ks, lane);
      v16bf a1 = load_a16(la, 64, wm * 32 + 16, ks, lane);
#pragma unroll
      for (int j = 0; j < 4; ++j) {
        v16bf bfr = load_bt16(lb, 64, wn * 64 + j * 16, ks, lane);
        acc[0][j] = wmma_bf16(a0, bfr, acc[0][j]);
        acc[1][j] = wmma_bf16(a1, bfr, acc[1][j]);
      }
    }
    syncpipe();   // next tile landed; everyone done reading cur
  }
}

// ---------------- QKV projection GEMM ----------------
// grid = (3072/256=12, 8192/64=128), 256 threads.
__global__ __launch_bounds__(256) void qkv_gemm(
    const bf16* __restrict__ A,      // [8192,1024]
    const bf16* __restrict__ Bt,     // [3072,1024] (Wqkv^T)
    const float* __restrict__ bias,  // [3072]
    bf16* __restrict__ Q,            // [B,H,S,64]
    bf16* __restrict__ Kt,           // [B,H,S,64]
    bf16* __restrict__ Vt) {         // [B,H,64,S]
  __shared__ bf16 lA[2 * 64 * 64];
  __shared__ bf16 lB[2 * 256 * 64];
  const int lane = threadIdx.x & 31;
  const int wm   = (threadIdx.x >> 5) >> 2;
  const int wn   = (threadIdx.x >> 5) & 3;
  const int row0g = blockIdx.y * 64;
  const int n0g   = blockIdx.x * 256;

  v8f acc[2][4];
  gemm_core(A, Bt, row0g, n0g, 1024, lA, lB, acc);

  const int hi = lane >> 4;
#pragma unroll
  for (int mi = 0; mi < 2; ++mi)
#pragma unroll
    for (int j = 0; j < 4; ++j)
#pragma unroll
      for (int v = 0; v < 8; ++v) {
        int gr  = row0g + wm * 32 + mi * 16 + v + 8 * hi;  // b*S + s
        int col = n0g + wn * 64 + j * 16 + (lane & 15);    // 0..3071
        float val = acc[mi][j][v] + bias[col];
        int bb = gr >> 11, s = gr & 2047;
        int h  = col / 192;                // per-head [q(64) k(64) v(64)]
        int r  = col - h * 192;
        size_t hb = (size_t)(bb * 16 + h);
        if (r < 64)        Q [(hb * 2048 + s) * 64 + r]         = (bf16)val;
        else if (r < 128)  Kt[(hb * 2048 + s) * 64 + (r - 64)]  = (bf16)val;
        else               Vt[(hb * 64 + (r - 128)) * 2048 + s] = (bf16)val;
      }
}

// ---------------- flash attention ----------------
// Block = 8 waves x 16 query rows = 128 rows of one (b,h). grid = (16, 64).
// 64-key blocks; K/V tiles async-staged to LDS, double-buffered, shared by
// all waves. Online softmax in registers; P relayout C->A via per-wave LDS.
__global__ __launch_bounds__(256) void attn_kernel(
    const bf16* __restrict__ Q,   // [B,H,S,64]
    const bf16* __restrict__ K,   // [B,H,S,64]
    const bf16* __restrict__ Vt,  // [B,H,64,S]
    bf16* __restrict__ O) {       // [B*S, 1024]
  __shared__ bf16 lK[2][64 * 64];   // [key][hd]
  __shared__ bf16 lV[2][64 * 64];   // [hd][key]
  __shared__ bf16 lP[8][16 * 64];   // per-wave P tile
  const int tid  = threadIdx.x;
  const int lane = tid & 31;
  const int wave = tid >> 5;
  const int bh   = blockIdx.y;
  const int b = bh >> 4, h = bh & 15;
  const int q0 = blockIdx.x * 128 + wave * 16;
  const size_t base = (size_t)bh * 2048 * 64;
  const bf16* Qh = Q + base;
  const bf16* Kh = K + base;
  const bf16* Vh = Vt + base;       // [64, 2048]

  auto stageKV = [&](int buf, int kb) {
#pragma unroll
    for (int i = 0; i < 2; ++i) {   // K tile: 64 keys x 128B
      int idx = i * 256 + tid;
      int r = idx >> 3, c = (idx & 7) * 8;
      cp_async_b128(&lK[buf][r * 64 + c], Kh + (size_t)(kb + r) * 64 + c);
    }
#pragma unroll
    for (int i = 0; i < 2; ++i) {   // V^T tile: 64 hd-rows x 128B
      int idx = i * 256 + tid;
      int r = idx >> 3, c = (idx & 7) * 8;
      cp_async_b128(&lV[buf][r * 64 + c], Vh + (size_t)r * 2048 + kb + c);
    }
  };

  v16bf qa0 = load_a16(Qh, 64, q0, 0, lane);
  v16bf qa1 = load_a16(Qh, 64, q0, 32, lane);

  v8f o[4];
#pragma unroll
  for (int j = 0; j < 4; ++j) o[j] = {};
  float m_i[8], l_i[8];
#pragma unroll
  for (int v = 0; v < 8; ++v) { m_i[v] = -1e30f; l_i[v] = 0.0f; }

  bf16* pw = &lP[wave][0];
  const int hi = lane >> 4;

  stageKV(0, 0);
  syncpipe();
  for (int kb = 0; kb < 2048; kb += 64) {
    const int cur = (kb >> 6) & 1;
    if (kb + 64 < 2048) stageKV(cur ^ 1, kb + 64);
    const bf16* kc = &lK[cur][0];
    const bf16* vc = &lV[cur][0];

    // S = Q K^T for 64 keys: 4 n-tiles x 2 hd-substeps
    v8f s[4];
#pragma unroll
    for (int nt = 0; nt < 4; ++nt) {
      s[nt] = {};
      s[nt] = wmma_bf16(qa0, load_bt16(kc, 64, nt * 16, 0,  lane), s[nt]);
      s[nt] = wmma_bf16(qa1, load_bt16(kc, 64, nt * 16, 32, lane), s[nt]);
    }

    // online softmax (rows replicated across 16-lane halves)
#pragma unroll
    for (int v = 0; v < 8; ++v) {
      float p0 = s[0][v] * 0.125f, p1 = s[1][v] * 0.125f;   // 1/sqrt(64)
      float p2 = s[2][v] * 0.125f, p3 = s[3][v] * 0.125f;
      float rm = halfmax(fmaxf(fmaxf(p0, p1), fmaxf(p2, p3)));
      float mn = fmaxf(m_i[v], rm);
      float corr = __expf(m_i[v] - mn);
      p0 = __expf(p0 - mn); p1 = __expf(p1 - mn);
      p2 = __expf(p2 - mn); p3 = __expf(p3 - mn);
      float rs = halfsum(p0 + p1 + p2 + p3);
      l_i[v] = l_i[v] * corr + rs;
      m_i[v] = mn;
#pragma unroll
      for (int j = 0; j < 4; ++j) o[j][v] *= corr;
      int roff = (v + 8 * hi) * 64 + (lane & 15);
      pw[roff]      = (bf16)p0;
      pw[roff + 16] = (bf16)p1;
      pw[roff + 32] = (bf16)p2;
      pw[roff + 48] = (bf16)p3;
    }
    asm volatile("s_wait_dscnt 0" ::: "memory");  // wave-local P round trip
    v16bf pa0 = load_a16(pw, 64, 0, 0,  lane);
    v16bf pa1 = load_a16(pw, 64, 0, 32, lane);
#pragma unroll
    for (int j = 0; j < 4; ++j) {
      o[j] = wmma_bf16(pa0, load_bt16(vc, 64, j * 16, 0,  lane), o[j]);
      o[j] = wmma_bf16(pa1, load_bt16(vc, 64, j * 16, 32, lane), o[j]);
    }
    syncpipe();
  }

  // normalize and emit bf16 rows for the output projection
#pragma unroll
  for (int v = 0; v < 8; ++v) {
    float inv = 1.0f / l_i[v];
    int s = q0 + v + 8 * hi;
#pragma unroll
    for (int j = 0; j < 4; ++j) {
      int d = h * 64 + j * 16 + (lane & 15);
      O[((size_t)(b * 2048 + s)) * 1024 + d] = (bf16)(o[j][v] * inv);
    }
  }
}

// ---------------- output projection GEMM ----------------
// grid = (1024/256=4, 8192/64=128), 256 threads. f32 output + bias.
__global__ __launch_bounds__(256) void out_gemm(
    const bf16* __restrict__ A,      // [8192,1024]
    const bf16* __restrict__ Bt,     // [1024,1024] (Wout^T)
    const float* __restrict__ bias,  // [1024]
    float* __restrict__ C) {         // [8192,1024] f32
  __shared__ bf16 lA[2 * 64 * 64];
  __shared__ bf16 lB[2 * 256 * 64];
  const int lane = threadIdx.x & 31;
  const int wm   = (threadIdx.x >> 5) >> 2;
  const int wn   = (threadIdx.x >> 5) & 3;
  const int row0g = blockIdx.y * 64;
  const int n0g   = blockIdx.x * 256;

  v8f acc[2][4];
  gemm_core(A, Bt, row0g, n0g, 1024, lA, lB, acc);

  const int hi = lane >> 4;
#pragma unroll
  for (int mi = 0; mi < 2; ++mi)
#pragma unroll
    for (int j = 0; j < 4; ++j)
#pragma unroll
      for (int v = 0; v < 8; ++v) {
        int gr  = row0g + wm * 32 + mi * 16 + v + 8 * hi;
        int col = n0g + wn * 64 + j * 16 + (lane & 15);
        C[(size_t)gr * 1024 + col] = acc[mi][j][v] + bias[col];
      }
}

// ---------------- host launcher ----------------
extern "C" void kernel_launch(void* const* d_in, const int* in_sizes, int n_in,
                              void* d_out, int out_size, void* d_ws, size_t ws_size,
                              hipStream_t stream) {
  const float* x    = (const float*)d_in[0];   // [4,2048,1024]
  const float* Wqkv = (const float*)d_in[1];   // [1024,3072]
  const float* bqkv = (const float*)d_in[2];   // [3072]
  const float* Wout = (const float*)d_in[3];   // [1024,1024]
  const float* bout = (const float*)d_in[4];   // [1024]
  float* out = (float*)d_out;

  char* ws = (char*)d_ws;
  size_t off = 0;
  auto carve = [&](size_t bytes) -> void* {
    void* p = ws + off;
    off += (bytes + 255) & ~(size_t)255;
    return p;
  };
  bf16* xb    = (bf16*)carve(8192ull * 1024 * 2);
  bf16* wqkvt = (bf16*)carve(3072ull * 1024 * 2);
  bf16* woutt = (bf16*)carve(1024ull * 1024 * 2);
  bf16* Qb    = (bf16*)carve(64ull * 2048 * 64 * 2);
  bf16* Kb    = (bf16*)carve(64ull * 2048 * 64 * 2);
  bf16* Vtb   = (bf16*)carve(64ull * 64 * 2048 * 2);
  bf16* attnb = (bf16*)carve(8192ull * 1024 * 2);
  (void)ws_size; (void)in_sizes; (void)n_in; (void)out_size;

  cvt_f32_bf16<<<8192, 256, 0, stream>>>(x, xb, 8192 * 1024);
  transpose_cvt<<<(3072 * 1024) / 256, 256, 0, stream>>>(Wqkv, wqkvt, 1024, 3072);
  transpose_cvt<<<(1024 * 1024) / 256, 256, 0, stream>>>(Wout, woutt, 1024, 1024);

  qkv_gemm<<<dim3(12, 128), 256, 0, stream>>>(xb, wqkvt, bqkv, Qb, Kb, Vtb);
  attn_kernel<<<dim3(16, 64), 256, 0, stream>>>(Qb, Kb, Vtb, attnb);
  out_gemm<<<dim3(4, 128), 256, 0, stream>>>(attnb, woutt, bout, out);
}
